// OptimizedSS2D_71047349010915
// MI455X (gfx1250) — matrix operand order repaired and checked by reading, not compiled, not run on hardware
//
#include <hip/hip_runtime.h>
#include <hip/hip_bf16.h>

// ---------------- problem constants ----------------
constexpr int D_MODEL = 192;
constexpr int D_STATE = 16;
constexpr int D_CONV  = 3;
constexpr int D_INNER = 384;
constexpr int DT_RANK = 12;
constexpr int BATCH   = 8;
constexpr int SEQLEN  = 4096;
constexpr int M_ROWS  = BATCH * SEQLEN;          // 32768
constexpr int M_TILES = M_ROWS / 16;             // 2048
constexpr int N_XZ    = 2 * D_INNER;             // 768
constexpr int N_XP    = 48;                      // 44 padded to 48
constexpr int MT      = 8;                       // m-tiles per wave (global-path GEMM)
constexpr int MT_BLK  = 8;                       // m-tiles per block (TDM-path GEMM)

typedef _Float16 v16h __attribute__((ext_vector_type(16)));
typedef _Float16 v8h  __attribute__((ext_vector_type(8)));
typedef float    v8f  __attribute__((ext_vector_type(8)));
typedef unsigned int uint32x4 __attribute__((ext_vector_type(4)));
typedef int          int32x8  __attribute__((ext_vector_type(8)));
typedef int          int32x4  __attribute__((ext_vector_type(4)));

#if defined(__has_builtin)
#if __has_builtin(__builtin_amdgcn_tensor_load_to_lds) && \
    __has_builtin(__builtin_amdgcn_s_wait_tensorcnt)
#define HAVE_TDM 1
#endif
#endif

// ---------------- f32 -> f16 conversion kernels ----------------
__global__ void cvt_f16_kernel(const float* __restrict__ src,
                               _Float16* __restrict__ dst, int n) {
    int i = blockIdx.x * blockDim.x + threadIdx.x;
    if (i < n) dst[i] = (_Float16)src[i];
}

// x_proj_w is (44, 384); pad rows 44..47 with zeros -> (48, 384) f16
__global__ void cvt_pad_xproj_kernel(const float* __restrict__ src,
                                     _Float16* __restrict__ dst) {
    int i = blockIdx.x * blockDim.x + threadIdx.x;
    if (i >= N_XP * D_INNER) return;
    int row = i / D_INNER;
    dst[i] = (row < (DT_RANK + 2 * D_STATE)) ? (_Float16)src[i] : (_Float16)0.0f;
}

// ---------------- WMMA fragment helpers (CDNA5 ISA 7.12.2, wave32) ----------------
// B 32x16 f16: lane n (mod 16) = column n; lanes 0-15: elems 0-7 -> K k0..k0+7,
//              elems 8-15 -> k0+16..k0+23; lanes 16-31: k0+8..k0+15, k0+24..k0+31.
template <int KCHUNKS>
__device__ __forceinline__ void load_b_frags(const _Float16* __restrict__ W,
                                             int ldw, v16h (&bf)[KCHUNKS]) {
    const int lane = threadIdx.x & 31;
    const _Float16* brow =
        W + (size_t)(lane & 15) * ldw + (lane >> 4) * 8;
#pragma unroll
    for (int c = 0; c < KCHUNKS; ++c) {
        v8h lo = *(const v8h*)(brow + c * 32);
        v8h hi = *(const v8h*)(brow + c * 32 + 16);
#pragma unroll
        for (int i = 0; i < 8; ++i) { bf[c][i] = lo[i]; bf[c][i + 8] = hi[i]; }
    }
}

// A 16x32 f16: lanes 0-15 = rows, elems 0-7 -> K k0..k0+7, elems 8-15 -> k0+16..k0+23;
//              lanes 16-31: elems 0-7 -> k0+8..k0+15, elems 8-15 -> k0+24..k0+31.
// Simple fused form (global-path kernels; compiler pipelines across tiles).
template <int KCHUNKS>
__device__ __forceinline__ v8f mma_rowtile(const _Float16* A, int lda,
                                           const v16h (&bf)[KCHUNKS]) {
    const int lane = threadIdx.x & 31;
    const _Float16* arow = A + (size_t)(lane & 15) * lda + (lane >> 4) * 8;
    v8f acc = {};
#pragma unroll
    for (int c = 0; c < KCHUNKS; ++c) {
        v8h lo = *(const v8h*)(arow + c * 32);
        v8h hi = *(const v8h*)(arow + c * 32 + 16);
        v16h a;
#pragma unroll
        for (int i = 0; i < 8; ++i) { a[i] = lo[i]; a[i + 8] = hi[i]; }
        acc = __builtin_amdgcn_wmma_f32_16x16x32_f16(
            false, a, false, bf[c], (short)0, acc, false, false);
    }
    return acc;
}

// Grouped two-phase form (LDS-path kernels): issue GROUP ds_loads, one wait,
// then a back-to-back WMMA chain (co-executes with the next group's loads).
template <int KCHUNKS, int GROUP>
__device__ __forceinline__ v8f mma_rowtile_grouped(const _Float16* A, int lda,
                                                   const v16h (&bf)[KCHUNKS]) {
    static_assert(KCHUNKS % GROUP == 0, "GROUP must divide KCHUNKS");
    const int lane = threadIdx.x & 31;
    const _Float16* arow = A + (size_t)(lane & 15) * lda + (lane >> 4) * 8;
    v8f acc = {};
#pragma unroll
    for (int g = 0; g < KCHUNKS; g += GROUP) {
        v16h a[GROUP];
#pragma unroll
        for (int c = 0; c < GROUP; ++c) {
            v8h lo = *(const v8h*)(arow + (g + c) * 32);
            v8h hi = *(const v8h*)(arow + (g + c) * 32 + 16);
#pragma unroll
            for (int i = 0; i < 8; ++i) { a[c][i] = lo[i]; a[c][i + 8] = hi[i]; }
        }
#pragma unroll
        for (int c = 0; c < GROUP; ++c)
            acc = __builtin_amdgcn_wmma_f32_16x16x32_f16(
                false, a[c], false, bf[g + c], (short)0, acc, false, false);
    }
    return acc;
}

// ---------------- Tensor Data Mover: stage a 16 x K f16 tile into LDS ----------------
// D# per CDNA5 ISA 8.3/8.4: group0 = {count/lds_addr/global_addr/type},
// group1 = {data_size, tensor dims/strides, tile dims}.  2D tile:
// tile_dim0 = K elems (row), tile_dim1 = 16 rows, row stride = K elems.
// This toolchain exposes the 6-arg builtin:
//   (uint32x4 g0, int32x8 g1, int32x4 g2, int32x4 g3, int32x8 g4, i32 cpol)
__device__ __forceinline__ unsigned lds_addr_of(const void* p) {
    // LDS aperture: low 32 bits of the flat shared address are the LDS offset
    return (unsigned)(unsigned long long)(uintptr_t)p;
}

__device__ __forceinline__ void tdm_load_tile16xK(void* dst_lds,
                                                  const _Float16* src,
                                                  unsigned kelems) {
#if defined(HAVE_TDM)
    unsigned long long ga = (unsigned long long)(uintptr_t)src;
    unsigned td0 = kelems;              // tensor_dim0 (elements)
    unsigned td1 = 1u << 20;            // tensor_dim1: large (no OOB clipping)
    uint32x4 g0;
    g0[0] = 1u;                                          // count=1, user mode
    g0[1] = lds_addr_of(dst_lds);                        // lds_addr [63:32]
    g0[2] = (unsigned)(ga & 0xFFFFFFFFu);                // global_addr[31:0]
    g0[3] = (unsigned)((ga >> 32) & 0x01FFFFFFu)         // global_addr[56:32]
          | (2u << 30);                                  // type=2 ("image")
    int32x8 g1;
    g1[0] = (int)(1u << 16);                             // data_size=1 -> 2 bytes
    g1[1] = (int)((td0 & 0xFFFFu) << 16);                // tensor_dim0[15:0] @ [63:48]
    g1[2] = (int)((td0 >> 16) | ((td1 & 0xFFFFu) << 16));// dim0 hi | dim1 lo
    g1[3] = (int)((td1 >> 16) | ((kelems & 0xFFFFu) << 16)); // dim1 hi | tile_dim0
    g1[4] = (int)(16u);                                  // tile_dim1=16, tile_dim2=0
    g1[5] = (int)kelems;                                 // tensor_dim0_stride[31:0]
    g1[6] = 0;                                           // stride hi | dim1_stride lo
    g1[7] = 0;
    int32x4 zz4 = {0, 0, 0, 0};
    int32x8 zz8 = {0, 0, 0, 0, 0, 0, 0, 0};
    __builtin_amdgcn_tensor_load_to_lds(g0, g1, zz4, zz4, zz8, 0);
#else
    (void)dst_lds; (void)src; (void)kelems;
#endif
}

// D-matrix layout: VGPR r -> M = r + 8*(lane>=16), N = lane & 15.

// ---------------- GEMM 1 (TDM path): xz = hs @ in_proj_w^T ----------------
// block = 256 (8 waves), grid = (M_TILES/MT_BLK, 48/8).
// Each wave owns one n-tile (weights register-resident); the block shares a
// double-buffered LDS A-tile streamed by the TDM (issued by wave 0).
__global__ void gemm_xz_kernel(const _Float16* __restrict__ hs16,
                               const _Float16* __restrict__ w16,
                               float* __restrict__ x, float* __restrict__ z) {
    __shared__ _Float16 sA[2][16 * D_MODEL];             // 2 x 6 KB
    const int wave  = threadIdx.x >> 5;
    const int lane  = threadIdx.x & 31;
    const int ntile = blockIdx.y * 8 + wave;
    const int mg    = blockIdx.x;

    v16h bf[D_MODEL / 32];                               // 6 frags in VGPRs
    load_b_frags<D_MODEL / 32>(w16 + (size_t)ntile * 16 * D_MODEL, D_MODEL, bf);

    const int hi  = lane >> 4;
    const int n   = ntile * 16 + (lane & 15);
    float* dst    = (n < D_INNER) ? x : z;
    const int col = (n < D_INNER) ? n : n - D_INNER;

#if defined(HAVE_TDM)
    if (wave == 0)   // TDM ignores EXEC: one issue per wave -> gate by wave
        tdm_load_tile16xK(&sA[0][0],
                          hs16 + (size_t)(mg * MT_BLK) * 16 * D_MODEL, D_MODEL);
#endif
    for (int t = 0; t < MT_BLK; ++t) {
        const int cur   = t & 1;
        const int mtile = mg * MT_BLK + t;
        __syncthreads();                       // readers of sA[cur^1] are done
#if defined(HAVE_TDM)
        if (wave == 0) {
            if (t + 1 < MT_BLK) {
                tdm_load_tile16xK(&sA[cur ^ 1][0],
                                  hs16 + (size_t)(mtile + 1) * 16 * D_MODEL,
                                  D_MODEL);
                __builtin_amdgcn_s_wait_tensorcnt((short)1);  // tile t done
            } else {
                __builtin_amdgcn_s_wait_tensorcnt((short)0);
            }
        }
#else
        for (int i = threadIdx.x; i < 16 * D_MODEL / 2; i += 256)
            ((unsigned*)&sA[cur][0])[i] =
                ((const unsigned*)(hs16 + (size_t)mtile * 16 * D_MODEL))[i];
#endif
        __syncthreads();                       // sA[cur] visible to all waves
        v8f acc = mma_rowtile_grouped<D_MODEL / 32, D_MODEL / 32>(
            &sA[cur][0], D_MODEL, bf);         // stage all 6 chunks, then 6 WMMAs
#pragma unroll
        for (int r = 0; r < 8; ++r)
            dst[(size_t)(mtile * 16 + r + 8 * hi) * D_INNER + col] = acc[r];
    }
}

// ---------------- depthwise conv (k=3, pad=1) + SiLU ----------------
__global__ void conv_silu_kernel(const float* __restrict__ x,
                                 const float* __restrict__ cw,
                                 const float* __restrict__ cb,
                                 float* __restrict__ u,
                                 _Float16* __restrict__ u16) {
    int idx = blockIdx.x * blockDim.x + threadIdx.x;
    if (idx >= M_ROWS * D_INNER) return;
    int d = idx % D_INNER;
    int m = idx / D_INNER;            // m = b*L + l
    int l = m % SEQLEN;
    float w0 = cw[d * 3 + 0], w1 = cw[d * 3 + 1], w2 = cw[d * 3 + 2];
    float xm = (l > 0)          ? x[(size_t)(m - 1) * D_INNER + d] : 0.0f;
    float xc = x[(size_t)m * D_INNER + d];
    float xp = (l < SEQLEN - 1) ? x[(size_t)(m + 1) * D_INNER + d] : 0.0f;
    float a  = cb[d] + xm * w0 + xc * w1 + xp * w2;
    float s  = a / (1.0f + __expf(-a));   // silu
    u[idx]   = s;
    u16[idx] = (_Float16)s;
}

// ---------------- GEMM 2 (global path): x_dbl = u @ x_proj_w^T ----------------
// Only 3 n-tiles; keep the register-B / global-A sweep (compiler pipelines it).
__global__ void gemm_xdbl_kernel(const _Float16* __restrict__ u16,
                                 const _Float16* __restrict__ wxp16,
                                 float* __restrict__ dtr,
                                 float* __restrict__ Bc,
                                 float* __restrict__ Cc) {
    const int wave  = threadIdx.x >> 5;
    const int mg    = blockIdx.x * 8 + wave;
    const int ntile = blockIdx.y;
    v16h bf[D_INNER / 32];                       // 12 frags
    load_b_frags<D_INNER / 32>(wxp16 + (size_t)ntile * 16 * D_INNER, D_INNER, bf);

    const int lane = threadIdx.x & 31;
    const int hi   = lane >> 4;
    const int n    = ntile * 16 + (lane & 15);
#pragma unroll
    for (int t = 0; t < MT; ++t) {
        const int mtile = mg * MT + t;
        v8f acc = mma_rowtile<D_INNER / 32>(
            u16 + (size_t)mtile * 16 * D_INNER, D_INNER, bf);
#pragma unroll
        for (int r = 0; r < 8; ++r) {
            int m = mtile * 16 + r + 8 * hi;
            float v = acc[r];
            if (n < DT_RANK)                    dtr[(size_t)m * DT_RANK + n] = v;
            else if (n < DT_RANK + D_STATE)     Bc[(size_t)m * D_STATE + (n - DT_RANK)] = v;
            else if (n < DT_RANK + 2 * D_STATE) Cc[(size_t)m * D_STATE + (n - DT_RANK - D_STATE)] = v;
        }
    }
}

// ---------------- dt = softplus(dtr @ dt_proj_w^T + 2*dt_proj_b) ----------------
// (the reference adds dt_proj_b twice; reproduced faithfully)
__global__ void dt_softplus_kernel(const float* __restrict__ dtr,
                                   const float* __restrict__ dtw,
                                   const float* __restrict__ dtb,
                                   float* __restrict__ dt) {
    int idx = blockIdx.x * blockDim.x + threadIdx.x;
    if (idx >= M_ROWS * D_INNER) return;
    int d = idx % D_INNER;
    int m = idx / D_INNER;
    float acc = 2.0f * dtb[d];
#pragma unroll
    for (int r = 0; r < DT_RANK; ++r)
        acc = fmaf(dtr[(size_t)m * DT_RANK + r], dtw[d * DT_RANK + r], acc);
    dt[idx] = (acc > 20.0f) ? acc : log1pf(__expf(acc));
}

// ---------------- selective scan (sequential in L) ----------------
constexpr int SCHUNK = 16;

__global__ void scan_kernel(const float* __restrict__ dt,
                            const float* __restrict__ u,
                            const float* __restrict__ Bc,
                            const float* __restrict__ Cc,
                            const float* __restrict__ z,
                            const float* __restrict__ A_log,
                            const float* __restrict__ D_skip,
                            _Float16* __restrict__ y16) {
    const int d = blockIdx.x * blockDim.x + threadIdx.x;   // 0..383
    const int b = blockIdx.y;
    float A[D_STATE], h[D_STATE];
#pragma unroll
    for (int n = 0; n < D_STATE; ++n) {
        A[n] = -__expf(A_log[d * D_STATE + n]);
        h[n] = 0.0f;
    }
    const float Dd = D_skip[d];
    __shared__ float sB[SCHUNK][D_STATE];
    __shared__ float sC[SCHUNK][D_STATE];
    const size_t base = (size_t)b * SEQLEN;

    for (int l0 = 0; l0 < SEQLEN; l0 += SCHUNK) {
        __syncthreads();   // previous chunk fully consumed
#pragma unroll
        for (int i = 0; i < (SCHUNK * 2 * D_STATE) / 128; ++i) {
            int pos = threadIdx.x + i * 128;       // 0..511
            int ll  = pos >> 5;
            int j   = pos & 31;
            size_t m = base + l0 + ll;
            if (j < D_STATE) sB[ll][j]           = Bc[m * D_STATE + j];
            else             sC[ll][j - D_STATE] = Cc[m * D_STATE + (j - D_STATE)];
        }
        __syncthreads();
        if (l0 + SCHUNK < SEQLEN) {
            size_t mn = (base + l0 + SCHUNK) * D_INNER + d;
            __builtin_prefetch(&dt[mn], 0, 1);
            __builtin_prefetch(&u[mn], 0, 1);
            __builtin_prefetch(&z[mn], 0, 1);
        }
        for (int lt = 0; lt < SCHUNK; ++lt) {
            size_t m  = base + l0 + lt;
            float dti = dt[m * D_INNER + d];
            float ui  = u[m * D_INNER + d];
            float du  = dti * ui;
            float y   = Dd * ui;
#pragma unroll
            for (int n = 0; n < D_STATE; ++n) {
                float ad = __expf(dti * A[n]);
                h[n] = fmaf(ad, h[n], du * sB[lt][n]);
                y    = fmaf(h[n], sC[lt][n], y);
            }
            float zi = z[m * D_INNER + d];
            float g  = zi / (1.0f + __expf(-zi));        // silu(z)
            y16[m * D_INNER + d] = (_Float16)(y * g);
        }
    }
}

// ---------------- GEMM 3 (TDM path): out = y @ out_proj_w^T ----------------
// block = 128 (4 waves), grid = (M_TILES/MT_BLK, 12/4).
__global__ void gemm_out_kernel(const _Float16* __restrict__ y16,
                                const _Float16* __restrict__ wout16,
                                float* __restrict__ out) {
    __shared__ _Float16 sA[2][16 * D_INNER];             // 2 x 12 KB
    const int wave  = threadIdx.x >> 5;
    const int lane  = threadIdx.x & 31;
    const int ntile = blockIdx.y * 4 + wave;
    const int mg    = blockIdx.x;

    v16h bf[D_INNER / 32];                               // 12 frags
    load_b_frags<D_INNER / 32>(wout16 + (size_t)ntile * 16 * D_INNER, D_INNER, bf);

    const int hi = lane >> 4;
    const int n  = ntile * 16 + (lane & 15);

#if defined(HAVE_TDM)
    if (wave == 0)
        tdm_load_tile16xK(&sA[0][0],
                          y16 + (size_t)(mg * MT_BLK) * 16 * D_INNER, D_INNER);
#endif
    for (int t = 0; t < MT_BLK; ++t) {
        const int cur   = t & 1;
        const int mtile = mg * MT_BLK + t;
        __syncthreads();
#if defined(HAVE_TDM)
        if (wave == 0) {
            if (t + 1 < MT_BLK) {
                tdm_load_tile16xK(&sA[cur ^ 1][0],
                                  y16 + (size_t)(mtile + 1) * 16 * D_INNER,
                                  D_INNER);
                __builtin_amdgcn_s_wait_tensorcnt((short)1);
            } else {
                __builtin_amdgcn_s_wait_tensorcnt((short)0);
            }
        }
#else
        for (int i = threadIdx.x; i < 16 * D_INNER / 2; i += 128)
            ((unsigned*)&sA[cur][0])[i] =
                ((const unsigned*)(y16 + (size_t)mtile * 16 * D_INNER))[i];
#endif
        __syncthreads();
        v8f acc = mma_rowtile_grouped<D_INNER / 32, 6>(  // 2 groups of 6
            &sA[cur][0], D_INNER, bf);
#pragma unroll
        for (int r = 0; r < 8; ++r)
            out[(size_t)(mtile * 16 + r + 8 * hi) * D_MODEL + n] = acc[r];
    }
}

// ---------------- host-side launch ----------------
extern "C" void kernel_launch(void* const* d_in, const int* in_sizes, int n_in,
                              void* d_out, int out_size, void* d_ws, size_t ws_size,
                              hipStream_t stream) {
    const float* hs     = (const float*)d_in[0];   // (8, 4096, 192)
    const float* w_in   = (const float*)d_in[1];   // (768, 192)
    const float* conv_w = (const float*)d_in[2];   // (384, 1, 3)
    const float* conv_b = (const float*)d_in[3];   // (384,)
    const float* w_xp   = (const float*)d_in[4];   // (44, 384)
    const float* dt_w   = (const float*)d_in[5];   // (384, 12)
    const float* dt_b   = (const float*)d_in[6];   // (384,)
    const float* A_log  = (const float*)d_in[7];   // (384, 16)
    const float* D_skip = (const float*)d_in[8];   // (384,)
    const float* w_out  = (const float*)d_in[9];   // (192, 384)
    float* out = (float*)d_out;                    // (8, 4096, 192) f32

    char* p = (char*)d_ws;
    auto alloc = [&](size_t bytes) -> void* {
        void* r = (void*)p;
        p += (bytes + 255) & ~(size_t)255;
        return r;
    };
    _Float16* hs16   = (_Float16*)alloc((size_t)M_ROWS * D_MODEL * 2);
    _Float16* win16  = (_Float16*)alloc((size_t)N_XZ * D_MODEL * 2);
    _Float16* wxp16  = (_Float16*)alloc((size_t)N_XP * D_INNER * 2);
    _Float16* wout16 = (_Float16*)alloc((size_t)D_MODEL * D_INNER * 2);
    float*    x_buf  = (float*)alloc((size_t)M_ROWS * D_INNER * 4);
    float*    z_buf  = (float*)alloc((size_t)M_ROWS * D_INNER * 4);
    float*    u_buf  = (float*)alloc((size_t)M_ROWS * D_INNER * 4);
    _Float16* u16    = (_Float16*)alloc((size_t)M_ROWS * D_INNER * 2);
    float*    dtr    = (float*)alloc((size_t)M_ROWS * DT_RANK * 4);
    float*    B_buf  = (float*)alloc((size_t)M_ROWS * D_STATE * 4);
    float*    C_buf  = (float*)alloc((size_t)M_ROWS * D_STATE * 4);
    float*    dt_buf = (float*)alloc((size_t)M_ROWS * D_INNER * 4);
    _Float16* y16    = (_Float16*)alloc((size_t)M_ROWS * D_INNER * 2);

    // --- precision conversions ---
    {
        int n = M_ROWS * D_MODEL;
        cvt_f16_kernel<<<(n + 255) / 256, 256, 0, stream>>>(hs, hs16, n);
        n = N_XZ * D_MODEL;
        cvt_f16_kernel<<<(n + 255) / 256, 256, 0, stream>>>(w_in, win16, n);
        n = N_XP * D_INNER;
        cvt_pad_xproj_kernel<<<(n + 255) / 256, 256, 0, stream>>>(w_xp, wxp16);
        n = D_MODEL * D_INNER;
        cvt_f16_kernel<<<(n + 255) / 256, 256, 0, stream>>>(w_out, wout16, n);
    }

    // --- GEMM 1: in_proj -> x, z (TDM-staged A, 8 n-tiles per block) ---
    gemm_xz_kernel<<<dim3(M_TILES / MT_BLK, N_XZ / 16 / 8), 256, 0, stream>>>(
        hs16, win16, x_buf, z_buf);

    // --- conv + silu ---
    {
        int n = M_ROWS * D_INNER;
        conv_silu_kernel<<<(n + 255) / 256, 256, 0, stream>>>(
            x_buf, conv_w, conv_b, u_buf, u16);
    }

    // --- GEMM 2: x_proj -> dtr, B, C ---
    gemm_xdbl_kernel<<<dim3(M_TILES / (8 * MT), N_XP / 16), 256, 0, stream>>>(
        u16, wxp16, dtr, B_buf, C_buf);

    // --- dt_proj + softplus ---
    {
        int n = M_ROWS * D_INNER;
        dt_softplus_kernel<<<(n + 255) / 256, 256, 0, stream>>>(
            dtr, dt_w, dt_b, dt_buf);
    }

    // --- selective scan + silu(z) gating -> y16 ---
    scan_kernel<<<dim3(D_INNER / 128, BATCH), 128, 0, stream>>>(
        dt_buf, u_buf, B_buf, C_buf, z_buf, A_log, D_skip, y16);

    // --- GEMM 3: out_proj -> out (TDM-staged A, 4 n-tiles per block) ---
    gemm_out_kernel<<<dim3(M_TILES / MT_BLK, D_MODEL / 16 / 4), 128, 0, stream>>>(
        y16, wout16, out);
}
